// wloss_72009421685329
// MI455X (gfx1250) — matrix operand order repaired and checked by reading, not compiled, not run on hardware
//
#include <hip/hip_runtime.h>
#include <hip/hip_bf16.h>
#include <math.h>

// ---------------------------------------------------------------------------
// Sinkhorn wloss on MI455X (gfx1250), wave32 + v_wmma_f32_16x16x32_bf16.
//
// Math: u0=1; 100x { v = b/(u@K); u = a_norm/(v@K^T) }; f = 10*log(u);
// loss = sum(a * (f/count - S/count^2)),  S = sum(a*f, axis=1), count = rowsum(a).
// ---------------------------------------------------------------------------

typedef __bf16 bf16_t;
typedef bf16_t v16bf __attribute__((ext_vector_type(16)));
typedef bf16_t v8bf  __attribute__((ext_vector_type(8)));
typedef float  v8f   __attribute__((ext_vector_type(8)));

#define BATCH 16
#define NDIM  4096            // n == m == 4096
#define KSEG  4               // K-dimension split for parallelism
#define REGC  10.0f

// float -> bf16 (round to nearest even), no __bf16 scalar arithmetic needed
__device__ __forceinline__ unsigned short f2bf(float f) {
    unsigned int u = __float_as_uint(f);
    unsigned int r = (u + 0x7FFFu + ((u >> 16) & 1u)) >> 16;
    return (unsigned short)r;
}

// ---------------------------------------------------------------------------
// One-time: K (f32 row-major [k][n]) -> Kb bf16 row-major, KTb bf16 transposed
// (KTb[n*4096+k] = K[k][n]).  Tiled 32x32 transpose through LDS.
// grid = 128*128 blocks of 256 threads.
// ---------------------------------------------------------------------------
__global__ void __launch_bounds__(256)
conv_k(const float* __restrict__ K, unsigned short* __restrict__ Kb,
       unsigned short* __restrict__ KTb) {
    __shared__ float tile[32][33];
    const int tx = threadIdx.x & 31;
    const int ty = threadIdx.x >> 5;
    const int bx = blockIdx.x & 127;
    const int by = blockIdx.x >> 7;
    const int x = bx * 32 + tx;
#pragma unroll
    for (int i = 0; i < 4; ++i) {
        int y = by * 32 + ty + i * 8;
        float v = K[(size_t)y * NDIM + x];
        Kb[(size_t)y * NDIM + x] = f2bf(v);
        tile[ty + i * 8][tx] = v;
    }
    __syncthreads();
#pragma unroll
    for (int i = 0; i < 4; ++i) {
        int yy = ty + i * 8;
        KTb[(size_t)(bx * 32 + yy) * NDIM + by * 32 + tx] = f2bf(tile[tx][yy]);
    }
}

// counts[b] = rowsum(a[b,:])   (16 blocks x 256)
__global__ void __launch_bounds__(256)
row_sum(const float* __restrict__ a, float* __restrict__ counts) {
    __shared__ float red[256];
    const int b = blockIdx.x;
    float s = 0.f;
    for (int i = threadIdx.x; i < NDIM; i += 256) s += a[b * NDIM + i];
    red[threadIdx.x] = s;
    __syncthreads();
    for (int off = 128; off; off >>= 1) {
        if (threadIdx.x < off) red[threadIdx.x] += red[threadIdx.x + off];
        __syncthreads();
    }
    if (threadIdx.x == 0) counts[b] = red[0];
}

// u = 1 (f32 + bf16)   (256 blocks x 256)
__global__ void __launch_bounds__(256)
init_u(float* __restrict__ uf, unsigned short* __restrict__ ub) {
    int i = blockIdx.x * 256 + threadIdx.x;
    uf[i] = 1.0f;
    ub[i] = 0x3F80u;  // bf16(1.0)
}

// ---------------------------------------------------------------------------
// Skinny GEMM: t_part[kseg][16][4096] = A(16x4096 bf16) @ B, where B is given
// in column-major bf16 storage Bcol (element (k,n) at Bcol[n*4096+k]).
//   - for t = u @ K     : Bcol = KTb  (KTb[n][k] == K[k][n])
//   - for s = v @ K^T   : Bcol = Kb   (Kb[i][j]  == K[i][j] == (K^T)(j,i) col-major)
// 64 blocks x 128 threads; each wave: one n-group of 4 tiles x one K segment.
// Per k-step: 1 A-tile load reused for 4 B tiles -> 4 WMMAs.
// ISA operand layouts (16-bit, wave32):
//   A 16x32 : lane=M%16, half=lane/16; elems0-7 = K[k0+8h .. +8), elems8-15 = K[k0+16+8h ..)
//   B 32x16 : N=lane%16; this code loads A and B alike with K=k0+16h..k0+16h+15 (not the ISA order)
//   C/D f32 : (M = r + 8*half, N = lane%16) in VGPR r
// ---------------------------------------------------------------------------
__global__ void __launch_bounds__(128)
sink_gemm(const unsigned short* __restrict__ A,
          const unsigned short* __restrict__ Bcol,
          float* __restrict__ tpart) {
    const int lane = threadIdx.x & 31;
    const int wave = blockIdx.x * 4 + (threadIdx.x >> 5);
    const int kseg = wave & 3;                 // 0..3
    const int ngrp = wave >> 2;                // 0..63
    const int half = lane >> 4;
    const int lidx = lane & 15;

    const int n0   = ngrp * 64;                // 4 tiles of 16 columns
    const int kbeg = kseg * (NDIM / KSEG);     // 1024-wide K segment
    const int aoff = half * 16;

    const unsigned short* aptr = A + (size_t)lidx * NDIM + kbeg;
    const unsigned short* bptr = Bcol + (size_t)(n0 + lidx) * NDIM + kbeg + half * 16;

    v8f acc0 = {}, acc1 = {}, acc2 = {}, acc3 = {};

    for (int k = 0; k < NDIM / KSEG; k += 32) {
        // A tile (shared by all 4 output tiles)
        v8bf alo = *(const v8bf*)(aptr + k + aoff);
        v8bf ahi = *(const v8bf*)(aptr + k + 8 + aoff);
        v16bf av = __builtin_shufflevector(alo, ahi,
                                           0, 1, 2, 3, 4, 5, 6, 7,
                                           8, 9, 10, 11, 12, 13, 14, 15);
        // 4 B tiles: 32 contiguous bytes per lane each (column-major storage)
        v16bf bv0 = *(const v16bf*)(bptr + 0 * 16 * NDIM + k);
        v16bf bv1 = *(const v16bf*)(bptr + 1 * 16 * NDIM + k);
        v16bf bv2 = *(const v16bf*)(bptr + 2 * 16 * NDIM + k);
        v16bf bv3 = *(const v16bf*)(bptr + 3 * 16 * NDIM + k);

        __builtin_prefetch(bptr + k + 256, 0, 1);   // global_prefetch_b8

        acc0 = __builtin_amdgcn_wmma_f32_16x16x32_bf16(false, av, false, bv0,
                                                       (short)0, acc0, false, false);
        acc1 = __builtin_amdgcn_wmma_f32_16x16x32_bf16(false, av, false, bv1,
                                                       (short)0, acc1, false, false);
        acc2 = __builtin_amdgcn_wmma_f32_16x16x32_bf16(false, av, false, bv2,
                                                       (short)0, acc2, false, false);
        acc3 = __builtin_amdgcn_wmma_f32_16x16x32_bf16(false, av, false, bv3,
                                                       (short)0, acc3, false, false);
    }

    float* out = tpart + (size_t)kseg * BATCH * NDIM;
#pragma unroll
    for (int r = 0; r < 8; ++r) {
        const int m = r + half * 8;
        out[(size_t)m * NDIM + n0 + 0 * 16 + lidx] = acc0[r];
        out[(size_t)m * NDIM + n0 + 1 * 16 + lidx] = acc1[r];
        out[(size_t)m * NDIM + n0 + 2 * 16 + lidx] = acc2[r];
        out[(size_t)m * NDIM + n0 + 3 * 16 + lidx] = acc3[r];
    }
}

// v = b / (sum of K-segment partials)          (256 blocks x 256)
__global__ void __launch_bounds__(256)
epi_v(const float* __restrict__ tpart, const float* __restrict__ bm,
      unsigned short* __restrict__ vb) {
    int i = blockIdx.x * 256 + threadIdx.x;
    float t = tpart[i] + tpart[i + 65536] + tpart[i + 2 * 65536] + tpart[i + 3 * 65536];
    vb[i] = f2bf(bm[i] / t);
}

// u = (a/count) / (sum of partials); keep f32 (for final log) + bf16
__global__ void __launch_bounds__(256)
epi_u(const float* __restrict__ tpart, const float* __restrict__ a,
      const float* __restrict__ counts, float* __restrict__ uf,
      unsigned short* __restrict__ ub) {
    int i = blockIdx.x * 256 + threadIdx.x;
    int m = i >> 12;
    float s = tpart[i] + tpart[i + 65536] + tpart[i + 2 * 65536] + tpart[i + 3 * 65536];
    float u = (a[i] / counts[m]) / s;
    uf[i] = u;
    ub[i] = f2bf(u);
}

// S[b] = sum_i a[b,i] * f[b,i],  f = REG*log(u)      (16 blocks x 256)
__global__ void __launch_bounds__(256)
calc_s(const float* __restrict__ a, const float* __restrict__ uf,
       float* __restrict__ S) {
    __shared__ float red[256];
    const int b = blockIdx.x;
    float s = 0.f;
    for (int i = threadIdx.x; i < NDIM; i += 256) {
        float f = REGC * __logf(uf[b * NDIM + i]);
        s += a[b * NDIM + i] * f;
    }
    red[threadIdx.x] = s;
    __syncthreads();
    for (int off = 128; off; off >>= 1) {
        if (threadIdx.x < off) red[threadIdx.x] += red[threadIdx.x + off];
        __syncthreads();
    }
    if (threadIdx.x == 0) S[b] = red[0];
}

// loss = sum a * (f/count - S/count^2)               (1 block x 256)
__global__ void __launch_bounds__(256)
calc_loss(const float* __restrict__ a, const float* __restrict__ uf,
          const float* __restrict__ counts, const float* __restrict__ S,
          float* __restrict__ out) {
    __shared__ float red[256];
    float acc = 0.f;
    for (int i = threadIdx.x; i < BATCH * NDIM; i += 256) {
        int m = i >> 12;
        float c = counts[m];
        float f = REGC * __logf(uf[i]);
        float img = f / c - S[m] / (c * c);
        acc += a[i] * img;
    }
    red[threadIdx.x] = acc;
    __syncthreads();
    for (int off = 128; off; off >>= 1) {
        if (threadIdx.x < off) red[threadIdx.x] += red[threadIdx.x + off];
        __syncthreads();
    }
    if (threadIdx.x == 0) out[0] = red[0];
}

// ---------------------------------------------------------------------------
extern "C" void kernel_launch(void* const* d_in, const int* in_sizes, int n_in,
                              void* d_out, int out_size, void* d_ws, size_t ws_size,
                              hipStream_t stream) {
    const float* a  = (const float*)d_in[0];   // [16,4096]
    const float* bm = (const float*)d_in[1];   // [16,4096]
    const float* K  = (const float*)d_in[2];   // [4096,4096]
    float* out = (float*)d_out;

    char* ws = (char*)d_ws;
    unsigned short* Kb     = (unsigned short*)(ws);                 // 32 MB
    unsigned short* KTb    = (unsigned short*)(ws + 33554432);      // 32 MB
    float*          uf     = (float*)(ws + 67108864);               // 256 KB
    unsigned short* ub     = (unsigned short*)(ws + 67371008);      // 128 KB
    unsigned short* vb     = (unsigned short*)(ws + 67502080);      // 128 KB
    float*          tpart  = (float*)(ws + 67633152);               // 1 MB
    float*          counts = (float*)(ws + 68681728);               // 64 B
    float*          S      = (float*)(ws + 68681792);               // 64 B

    conv_k<<<128 * 128, 256, 0, stream>>>(K, Kb, KTb);
    row_sum<<<BATCH, 256, 0, stream>>>(a, counts);
    init_u<<<256, 256, 0, stream>>>(uf, ub);

    for (int it = 0; it < 100; ++it) {
        sink_gemm<<<64, 128, 0, stream>>>(ub, KTb, tpart);          // t = u @ K
        epi_v<<<256, 256, 0, stream>>>(tpart, bm, vb);              // v = b / t
        sink_gemm<<<64, 128, 0, stream>>>(vb, Kb, tpart);           // s = v @ K^T
        epi_u<<<256, 256, 0, stream>>>(tpart, a, counts, uf, ub);   // u = a_n / s
    }

    calc_s<<<BATCH, 256, 0, stream>>>(a, uf, S);
    calc_loss<<<1, 256, 0, stream>>>(a, uf, counts, S, out);
}
